// KGSLomics_80633716015291
// MI455X (gfx1250) — matrix operand order repaired and checked by reading, not compiled, not run on hardware
//
#include <hip/hip_runtime.h>

// ---------------------------------------------------------------------------
// RGAT pipeline for gfx1250 (MI455X): f16 WMMA GEMMs + f16 edge gathers +
// f32 atomic scatters, with CDNA5 async data movement:
//   - TDM (tensor_load_to_lds) pulls the pre-swizzled B fragments into LDS
//     once per block (TENSORcnt + s_wait_tensorcnt).
//   - global_load_async_to_lds_b128 stages the A chunk (ASYNCcnt).
// Both are feature-gated with __has_builtin and fall back to the proven
// synchronous path.  The async builtin's signature (from the round-2
// diagnostic) is (global v4i*, shared v4i*, imm int offset, imm int cpol).
// ---------------------------------------------------------------------------

typedef __attribute__((ext_vector_type(16))) _Float16 v16h;
typedef __attribute__((ext_vector_type(8)))  _Float16 v8h;
typedef __attribute__((ext_vector_type(4)))  _Float16 v4h;
typedef __attribute__((ext_vector_type(8)))  float    v8f;
typedef unsigned int u32x4 __attribute__((ext_vector_type(4)));
typedef int          i32x8 __attribute__((ext_vector_type(8)));
typedef int          i32x4 __attribute__((ext_vector_type(4)));
typedef int          v4i   __attribute__((vector_size(16)));

#if defined(__has_builtin)
#  if __has_builtin(__builtin_amdgcn_tensor_load_to_lds)
#    define HAVE_TDM 1
#  endif
#  if __has_builtin(__builtin_amdgcn_global_load_async_to_lds_b128)
#    define HAVE_ASYNC_LDS 1
#  endif
#  if __has_builtin(__builtin_amdgcn_s_wait_asynccnt)
#    define HAVE_WAIT_ASYNC 1
#  endif
#endif
#ifndef HAVE_TDM
#define HAVE_TDM 0
#endif
#ifndef HAVE_ASYNC_LDS
#define HAVE_ASYNC_LDS 0
#endif
#ifndef HAVE_WAIT_ASYNC
#define HAVE_WAIT_ASYNC 0
#endif

// pointer casts matching the async builtin's parameter types
#define AS1_V4I(p) ((__attribute__((address_space(1))) v4i*)(p))
#define AS3_V4I(p) ((__attribute__((address_space(3))) v4i*)(p))

static constexpr int kN   = 50000;
static constexpr int kE   = 800000;
static constexpr int kR   = 8;
static constexpr int kH   = 4;
static constexpr int kCIN = 256;
static constexpr int kHID = 128;

// ---------------------------------------------------------------------------
// small helpers
// ---------------------------------------------------------------------------
__device__ __forceinline__ float leakyf(float v, float s) {
    return v > 0.0f ? v : s * v;
}

// ordered-int float atomic max (amax buffer initialized to a very negative value)
__device__ __forceinline__ void atomicMaxF(float* addr, float v) {
    if (v >= 0.0f) {
        atomicMax((int*)addr, __float_as_int(v));
    } else {
        atomicMin((unsigned int*)addr, __float_as_uint(v));
    }
}

#if HAVE_TDM
// 1-D LDS fill via the Tensor Data Mover: copies nHalves f16 elements from
// global (contiguous) to LDS (contiguous).  Descriptor per CDNA5 ISA:
// group0 = {count/lds_addr/global_addr/type}, group1 = dims/strides.
__device__ __forceinline__ void tdm_load_1d(const _Float16* gsrc,
                                            _Float16* ldsDst,
                                            unsigned nHalves) {
    unsigned long long ga = (unsigned long long)(uintptr_t)gsrc;
    unsigned ldsOff =
        (unsigned)(uintptr_t)(__attribute__((address_space(3))) _Float16*)ldsDst;
    u32x4 g0;
    g0.x = 1u;                                           // count=1, user mode
    g0.y = ldsOff;                                       // lds_addr (bytes)
    g0.z = (unsigned)(ga & 0xFFFFFFFFu);                 // global_addr[31:0]
    g0.w = (unsigned)((ga >> 32) & 0x01FFFFFFu) | (2u << 30);  // addr[56:32], type=2
    i32x8 g1;
    g1[0] = (1 << 16);                                   // data_size = 2 bytes
    g1[1] = (int)((nHalves & 0xFFFFu) << 16);            // tensor_dim0[15:0]
    g1[2] = (int)(((nHalves >> 16) & 0xFFFFu) | (1u << 16)); // dim0 hi | tensor_dim1=1
    g1[3] = (int)((nHalves & 0xFFFFu) << 16);            // tile_dim0
    g1[4] = 1;                                           // tile_dim1 = 1
    g1[5] = (int)nHalves;                                // tensor_dim0_stride lo
    g1[6] = 0;                                           // stride hi / dim1_stride lo
    g1[7] = 0;
    i32x4 z4 = {0, 0, 0, 0};
#if defined(__clang_major__) && (__clang_major__ >= 23)
    i32x8 z8 = {0, 0, 0, 0, 0, 0, 0, 0};
    __builtin_amdgcn_tensor_load_to_lds(g0, g1, z4, z4, z8, 0);
#else
    __builtin_amdgcn_tensor_load_to_lds(g0, g1, z4, z4, 0);
#endif
}
#endif

// ---------------------------------------------------------------------------
// fill a float buffer with a constant
// ---------------------------------------------------------------------------
__global__ void fill_f32(float* __restrict__ p, long long n, float v) {
    long long i = (long long)blockIdx.x * blockDim.x + threadIdx.x;
    if (i < n) p[i] = v;
}

// ---------------------------------------------------------------------------
// f32 -> f16 weight conversion, pre-swizzled into WMMA B-fragment layout.
// W: [R][K][128] row-major f32.
// Wf: [R][K/32][nt=8][lane=32][i=16] f16 (32 contiguous bytes per lane).
// B fragment order used here (contiguous k, A matches): lane holds col = lane%16,
// K-base = (lane<16 ? 0 : 16), element i -> K = Kbase + i.
// ---------------------------------------------------------------------------
__global__ void convert_wfrag(const float* __restrict__ W,
                              _Float16* __restrict__ Wf,
                              int K, int R) {
    int idx = blockIdx.x * 256 + threadIdx.x;       // one per (r, ch, nt, lane)
    int chunks = K >> 5;
    int total = R * chunks * 8 * 32;
    if (idx >= total) return;
    int lane = idx & 31;
    int nt   = (idx >> 5) & 7;
    int ch   = (idx >> 8) % chunks;
    int r    = idx / (chunks << 8);
    int col  = nt * 16 + (lane & 15);
    int kb   = ch * 32 + ((lane < 16) ? 0 : 16);
    const float* src = W + ((long long)r * K + kb) * 128 + col;
    _Float16* dst = Wf + (long long)idx * 16;
#pragma unroll
    for (int i = 0; i < 16; ++i) dst[i] = (_Float16)src[(long long)i * 128];
}

// ---------------------------------------------------------------------------
// CCLE 2-layer MLP + concat with kg embedding -> x_in (f16, [N,256]).
// ---------------------------------------------------------------------------
__global__ __launch_bounds__(128)
void ccle_concat(const float* __restrict__ kg,
                 const float* __restrict__ ccle,
                 const int* __restrict__ node_id,
                 const float* __restrict__ w1, const float* __restrict__ b1,
                 const float* __restrict__ w2, const float* __restrict__ b2,
                 _Float16* __restrict__ xinH) {
    __shared__ float h1[32];
    int n = blockIdx.x;
    int node = node_id[n];
    int t = threadIdx.x;
    if (t < 32) {
        float a = b1[t];
#pragma unroll
        for (int i = 0; i < 4; ++i) a += ccle[(long long)node * 4 + i] * w1[i * 32 + t];
        h1[t] = leakyf(a, 0.01f);
    }
    __syncthreads();
    float o = b2[t];
#pragma unroll
    for (int j = 0; j < 32; ++j) o += h1[j] * w2[j * 128 + t];
    long long base = (long long)n * 256;
    xinH[base + t]       = (_Float16)kg[(long long)node * 128 + t];
    xinH[base + 128 + t] = (_Float16)o;
}

// ---------------------------------------------------------------------------
// WMMA GEMM:  C[M,128] = A[M,K] (f16 row-major) @ Bfrag (pre-swizzled f16)
// + optional bias + leaky(slope)   (slope==1.0 -> identity)
// Block = 256 threads = 8 waves; each wave computes a 16x128 strip.
// Dynamic LDS:  [0,10240)   A chunk, fragment-swizzled, 40-half row stride
//               [10240,...) whole B fragment buffer for this relation
//                           (filled once by the TDM when available)
// ---------------------------------------------------------------------------
__global__ __launch_bounds__(256)
void wmma_gemm_nk128(const _Float16* __restrict__ A,
                     const _Float16* __restrict__ Bfrag,
                     int M, int K,
                     long long bRelStride, long long oRelStride,
                     const float* __restrict__ bias, float slope,
                     _Float16* __restrict__ outH, float* __restrict__ outF) {
    extern __shared__ __align__(16) char smem[];
    _Float16* ldsA = (_Float16*)smem;                   // 128*40 halves = 10240B
#if HAVE_TDM
    _Float16* ldsB = (_Float16*)(smem + 10240);         // (K/32)*4096 halves
#endif

    const int tid  = threadIdx.x;
    const int lane = tid & 31;
    const int w    = tid >> 5;
    const int m0   = blockIdx.x * 128;
    const int nChunks = K >> 5;
    const _Float16* Bp = Bfrag + (long long)blockIdx.y * bRelStride;

#if HAVE_TDM
    // One TDM op pulls the whole relation's B fragments (<=64KB) into LDS.
    if (tid == 0) tdm_load_1d(Bp, ldsB, (unsigned)(nChunks * 4096));
    __builtin_amdgcn_s_wait_tensorcnt(0);
    __syncthreads();
#endif

    v8f c[8] = {};

    const int sRow = tid >> 1;              // 0..127
    const int sKb  = (tid & 1) << 4;        // 0 or 16
    // row-local j order {k0-7, k8-15, k16-23, k24-31} (contiguous):
    const int j0 = sKb;
    const int j1 = sKb + 8;
    // clamp instead of zero-fill: rows >= M only feed C rows that are never
    // stored, and clamping keeps EXEC uniform for the async path.
    const int gRow = (m0 + sRow < M) ? (m0 + sRow) : (M - 1);

    for (int ch = 0; ch < nChunks; ++ch) {
        // ---- stage A chunk (128 rows x 32 halves) into swizzled LDS ----
        {
            const _Float16* ap = A + (long long)gRow * K + (ch << 5) + sKb;
#if HAVE_ASYNC_LDS
            __builtin_amdgcn_global_load_async_to_lds_b128(
                AS1_V4I(ap),     AS3_V4I(&ldsA[sRow * 40 + j0]), 0, 0);
            __builtin_amdgcn_global_load_async_to_lds_b128(
                AS1_V4I(ap + 8), AS3_V4I(&ldsA[sRow * 40 + j1]), 0, 0);
#if HAVE_WAIT_ASYNC
            __builtin_amdgcn_s_wait_asynccnt(0);
#else
            asm volatile("s_wait_asynccnt 0" ::: "memory");
#endif
#else
            v8h g0 = *(const v8h*)(ap);
            v8h g1 = *(const v8h*)(ap + 8);
            *(v8h*)(&ldsA[sRow * 40 + j0]) = g0;
            *(v8h*)(&ldsA[sRow * 40 + j1]) = g1;
#endif
        }
        __syncthreads();

        // ---- A fragment for this wave ----
        int r = w * 16 + (lane & 15);
        const v8h* ap2 = (const v8h*)(&ldsA[r * 40 + ((lane < 16) ? 0 : 16)]);
        v8h a0 = ap2[0];
        v8h a1 = ap2[1];
        v16h a = __builtin_shufflevector(a0, a1,
                 0, 1, 2, 3, 4, 5, 6, 7, 8, 9, 10, 11, 12, 13, 14, 15);

        // ---- B fragments (LDS when TDM filled it, else global), 8 WMMAs ----
#if HAVE_TDM
        const _Float16* bp = ldsB + ch * 4096 + lane * 16;
#else
        const _Float16* bp = Bp + (long long)ch * 4096 + lane * 16;
#endif
#pragma unroll
        for (int nt = 0; nt < 8; ++nt) {
            const v8h* bv = (const v8h*)(bp + nt * 512);
            v8h b0 = bv[0];
            v8h b1 = bv[1];
            v16h b = __builtin_shufflevector(b0, b1,
                     0, 1, 2, 3, 4, 5, 6, 7, 8, 9, 10, 11, 12, 13, 14, 15);
            c[nt] = __builtin_amdgcn_wmma_f32_16x16x32_f16(
                        false, a, false, b, (short)0, c[nt], false, false);
        }
        __syncthreads();
    }

    // ---- epilogue: bias + leaky, store f16 or f32 ----
    int rowBase = m0 + w * 16 + ((lane < 16) ? 0 : 8);
    int col0 = lane & 15;
    long long oRel = (long long)blockIdx.y * oRelStride;
#pragma unroll
    for (int nt = 0; nt < 8; ++nt) {
        int col = nt * 16 + col0;
        float bv = bias ? bias[col] : 0.0f;
#pragma unroll
        for (int i = 0; i < 8; ++i) {
            int row = rowBase + i;
            if (row < M) {
                float v = c[nt][i] + bv;
                v = v > 0.0f ? v : slope * v;
                long long idx = oRel + (long long)row * 128 + col;
                if (outH) outH[idx] = (_Float16)v;
                else      outF[idx] = v;
            }
        }
    }
}

// ---------------------------------------------------------------------------
// Edge attention logits: a[e,h] = leaky(out_i@q + out_j@k, 0.2),
// plus running per-destination segment max.  One wave per edge.
// ---------------------------------------------------------------------------
__global__ __launch_bounds__(256)
void edge_logits(const _Float16* __restrict__ hr,
                 const int* __restrict__ src, const int* __restrict__ dst,
                 const int* __restrict__ etype,
                 const float* __restrict__ q, const float* __restrict__ k,
                 float* __restrict__ elog, float* __restrict__ amax) {
    __shared__ float sq[128 * 4];
    __shared__ float sk[128 * 4];
    for (int i = threadIdx.x; i < 512; i += 256) { sq[i] = q[i]; sk[i] = k[i]; }
    __syncthreads();

    int e = blockIdx.x * 8 + (threadIdx.x >> 5);
    if (e >= kE) return;
    int lane = threadIdx.x & 31;
    int t = etype[e], s = src[e], d = dst[e];

    const _Float16* pj = hr + ((long long)t * kN + s) * 128 + lane * 4;
    const _Float16* pi = hr + ((long long)t * kN + d) * 128 + lane * 4;
    __builtin_prefetch(pj, 0, 1);           // row is re-gathered by the scatter pass
    v4h vj = *(const v4h*)pj;
    v4h vi = *(const v4h*)pi;

    float acc[4] = {0.f, 0.f, 0.f, 0.f};
#pragma unroll
    for (int cc = 0; cc < 4; ++cc) {
        int cidx = (lane * 4 + cc) * 4;
        float fi = (float)vi[cc];
        float fj = (float)vj[cc];
#pragma unroll
        for (int h = 0; h < 4; ++h) acc[h] += fi * sq[cidx + h] + fj * sk[cidx + h];
    }
#pragma unroll
    for (int off = 16; off > 0; off >>= 1) {
#pragma unroll
        for (int h = 0; h < 4; ++h) acc[h] += __shfl_xor(acc[h], off);
    }
    if (lane < 4) {
        float a = leakyf(acc[lane], 0.2f);
        elog[(long long)e * 4 + lane] = a;
        atomicMaxF(&amax[(long long)d * 4 + lane], a);
    }
}

// ---------------------------------------------------------------------------
// exp(a - amax[dst]) in place, plus per-destination denominator sum.
// ---------------------------------------------------------------------------
__global__ void edge_exp(const int* __restrict__ dst,
                         float* __restrict__ elog,
                         const float* __restrict__ amax,
                         float* __restrict__ denom) {
    long long idx = (long long)blockIdx.x * blockDim.x + threadIdx.x;
    if (idx >= (long long)kE * 4) return;
    int e = (int)(idx >> 2);
    int h = (int)(idx & 3);
    int d = dst[e];
    float ex = __expf(elog[idx] - amax[(long long)d * 4 + h]);
    elog[idx] = ex;
    atomicAdd(&denom[(long long)d * 4 + h], ex);
}

// ---------------------------------------------------------------------------
// Weighted message scatter: agg[dst] += alpha * out_j.  One wave per edge,
// 4 channels per lane, f32 global atomics.
// ---------------------------------------------------------------------------
__global__ __launch_bounds__(256)
void edge_scatter(const _Float16* __restrict__ hr,
                  const int* __restrict__ src, const int* __restrict__ dst,
                  const int* __restrict__ etype,
                  const float* __restrict__ elog,
                  const float* __restrict__ denom,
                  float* __restrict__ agg) {
    int e = blockIdx.x * 8 + (threadIdx.x >> 5);
    if (e >= kE) return;
    int lane = threadIdx.x & 31;
    int t = etype[e], s = src[e], d = dst[e];
    int head = lane >> 3;                   // channel block (lane*4)/32
    float alpha = elog[(long long)e * 4 + head] / denom[(long long)d * 4 + head];
    const _Float16* pj = hr + ((long long)t * kN + s) * 128 + lane * 4;
    v4h vj = *(const v4h*)pj;
    float* po = agg + (long long)d * 128 + lane * 4;
#pragma unroll
    for (int i = 0; i < 4; ++i) atomicAdd(&po[i], alpha * (float)vj[i]);
}

// ---------------------------------------------------------------------------
// x = leaky(agg + bias) -> f16   (layer-1 output, feeds layer-2 GEMM)
// ---------------------------------------------------------------------------
__global__ void combine_to_h(const float* __restrict__ agg,
                             const float* __restrict__ bias,
                             _Float16* __restrict__ outH, long long n) {
    long long i = (long long)blockIdx.x * blockDim.x + threadIdx.x;
    if (i >= n) return;
    int col = (int)(i & 127);
    outH[i] = (_Float16)leakyf(agg[i] + bias[col], 0.01f);
}

// ---------------------------------------------------------------------------
// out = leaky(agg + bias + skip) -> f32 (final output)
// ---------------------------------------------------------------------------
__global__ void combine_final(const float* __restrict__ agg,
                              const float* __restrict__ bias,
                              const float* __restrict__ skip,
                              float* __restrict__ out, long long n) {
    long long i = (long long)blockIdx.x * blockDim.x + threadIdx.x;
    if (i >= n) return;
    int col = (int)(i & 127);
    out[i] = leakyf(agg[i] + bias[col] + skip[i], 0.01f);
}

// ---------------------------------------------------------------------------
// host-side launcher
// ---------------------------------------------------------------------------
extern "C" void kernel_launch(void* const* d_in, const int* in_sizes, int n_in,
                              void* d_out, int out_size, void* d_ws, size_t ws_size,
                              hipStream_t stream) {
    const float* kg      = (const float*)d_in[0];
    const float* ccle    = (const float*)d_in[1];
    const int*   node_id = (const int*)d_in[2];
    const int*   eidx    = (const int*)d_in[3];
    const int*   etype   = (const int*)d_in[4];
    const float* cw1     = (const float*)d_in[5];
    const float* cb1     = (const float*)d_in[6];
    const float* cw2     = (const float*)d_in[7];
    const float* cb2     = (const float*)d_in[8];
    const float* W1      = (const float*)d_in[9];
    const float* q1      = (const float*)d_in[10];
    const float* k1      = (const float*)d_in[11];
    const float* b1      = (const float*)d_in[12];
    const float* W2      = (const float*)d_in[13];
    const float* q2      = (const float*)d_in[14];
    const float* k2      = (const float*)d_in[15];
    const float* b2      = (const float*)d_in[16];
    const float* sw1     = (const float*)d_in[17];
    const float* sb1     = (const float*)d_in[18];
    const float* sw2     = (const float*)d_in[19];
    const float* sb2     = (const float*)d_in[20];
    (void)in_sizes; (void)n_in; (void)out_size; (void)ws_size;

    const int* srcp = eidx;          // edge_index[0]
    const int* dstp = eidx + kE;     // edge_index[1]

    // ---- deterministic workspace carve-up ----
    char* wp = (char*)d_ws;
    auto alloc = [&](size_t bytes) -> void* {
        void* p = (void*)wp;
        wp += (bytes + 255) & ~(size_t)255;
        return p;
    };
    _Float16* xinH  = (_Float16*)alloc((size_t)kN * 256 * 2);
    _Float16* hrH   = (_Float16*)alloc((size_t)kR * kN * 128 * 2);
    float*    elog  = (float*)alloc((size_t)kE * kH * 4);
    float*    amax  = (float*)alloc((size_t)kN * kH * 4);
    float*    denom = (float*)alloc((size_t)kN * kH * 4);
    float*    agg   = (float*)alloc((size_t)kN * 128 * 4);
    _Float16* xH    = (_Float16*)alloc((size_t)kN * 128 * 2);
    _Float16* midH  = (_Float16*)alloc((size_t)kN * 128 * 2);
    float*    skipF = (float*)alloc((size_t)kN * 128 * 4);
    _Float16* W1f   = (_Float16*)alloc((size_t)kR * kCIN * 128 * 2);
    _Float16* W2f   = (_Float16*)alloc((size_t)kR * kHID * 128 * 2);
    _Float16* sw1f  = (_Float16*)alloc((size_t)kCIN * 128 * 2);
    _Float16* sw2f  = (_Float16*)alloc((size_t)kHID * 128 * 2);

    const int mTiles = (kN + 127) / 128;            // 391
    const long long nd = (long long)kN * 128;
    const int ndBlocks = (int)((nd + 255) / 256);
    const int nhBlocks = (int)(((long long)kN * kH + 255) / 256);
    const int eWaveBlocks = (kE + 7) / 8;
    const int ehBlocks = (int)(((long long)kE * 4 + 255) / 256);

    const size_t smemA    = 128 * 40 * 2;                      // 10240 B
    const size_t smemK256 = smemA + (kCIN / 32) * 4096 * 2;    // + 64 KB
    const size_t smemK128 = smemA + (kHID / 32) * 4096 * 2;    // + 32 KB

    // ---- weight conversion (f32 -> f16, WMMA B-fragment layout) ----
    convert_wfrag<<<(kR * (kCIN / 32) * 256 + 255) / 256, 256, 0, stream>>>(W1, W1f, kCIN, kR);
    convert_wfrag<<<(kR * (kHID / 32) * 256 + 255) / 256, 256, 0, stream>>>(W2, W2f, kHID, kR);
    convert_wfrag<<<((kCIN / 32) * 256 + 255) / 256, 256, 0, stream>>>(sw1, sw1f, kCIN, 1);
    convert_wfrag<<<((kHID / 32) * 256 + 255) / 256, 256, 0, stream>>>(sw2, sw2f, kHID, 1);

    // ---- encoder: ccle MLP + concat -> x_in (f16) ----
    ccle_concat<<<kN, 128, 0, stream>>>(kg, ccle, node_id, cw1, cb1, cw2, cb2, xinH);

    // ---- layer 1: hr = x_in @ W1[r]  (8 relations via grid.y) ----
    wmma_gemm_nk128<<<dim3(mTiles, kR), 256, smemK256, stream>>>(
        xinH, W1f, kN, kCIN,
        (long long)(kCIN / 32) * 4096, (long long)kN * 128,
        nullptr, 1.0f, hrH, nullptr);

    // attention round 1
    fill_f32<<<nhBlocks, 256, 0, stream>>>(amax, (long long)kN * kH, -1e30f);
    fill_f32<<<nhBlocks, 256, 0, stream>>>(denom, (long long)kN * kH, 0.0f);
    fill_f32<<<ndBlocks, 256, 0, stream>>>(agg, nd, 0.0f);
    edge_logits<<<eWaveBlocks, 256, 0, stream>>>(hrH, srcp, dstp, etype, q1, k1, elog, amax);
    edge_exp<<<ehBlocks, 256, 0, stream>>>(dstp, elog, amax, denom);
    edge_scatter<<<eWaveBlocks, 256, 0, stream>>>(hrH, srcp, dstp, etype, elog, denom, agg);
    combine_to_h<<<ndBlocks, 256, 0, stream>>>(agg, b1, xH, nd);

    // ---- layer 2: hr = x1 @ W2[r] ----
    wmma_gemm_nk128<<<dim3(mTiles, kR), 256, smemK128, stream>>>(
        xH, W2f, kN, kHID,
        (long long)(kHID / 32) * 4096, (long long)kN * 128,
        nullptr, 1.0f, hrH, nullptr);

    // attention round 2
    fill_f32<<<nhBlocks, 256, 0, stream>>>(amax, (long long)kN * kH, -1e30f);
    fill_f32<<<nhBlocks, 256, 0, stream>>>(denom, (long long)kN * kH, 0.0f);
    fill_f32<<<ndBlocks, 256, 0, stream>>>(agg, nd, 0.0f);
    edge_logits<<<eWaveBlocks, 256, 0, stream>>>(hrH, srcp, dstp, etype, q2, k2, elog, amax);
    edge_exp<<<ehBlocks, 256, 0, stream>>>(dstp, elog, amax, denom);
    edge_scatter<<<eWaveBlocks, 256, 0, stream>>>(hrH, srcp, dstp, etype, elog, denom, agg);

    // ---- skip path: leaky(x_in @ sw1 + sb1) @ sw2 + sb2 ----
    wmma_gemm_nk128<<<dim3(mTiles, 1), 256, smemK256, stream>>>(
        xinH, sw1f, kN, kCIN, 0, 0, sb1, 0.01f, midH, nullptr);
    wmma_gemm_nk128<<<dim3(mTiles, 1), 256, smemK128, stream>>>(
        midH, sw2f, kN, kHID, 0, 0, sb2, 1.0f, nullptr, skipF);

    // ---- final combine ----
    combine_final<<<ndBlocks, 256, 0, stream>>>(agg, b2, skipF, (float*)d_out, nd);
}
